// NAttention_68685116997998
// MI455X (gfx1250) — matrix operand order repaired and checked by reading, not compiled, not run on hardware
//
#include <hip/hip_runtime.h>
#include <hip/hip_bf16.h>

typedef _Float16 half_t;
typedef __attribute__((ext_vector_type(16))) _Float16 v16h;
typedef __attribute__((ext_vector_type(8)))  _Float16 v8h;
typedef __attribute__((ext_vector_type(8)))  float    v8f;

#define BH   56
#define BW   56
#define NHEADS 8
#define HD   64
#define KS   7
#define UW   22      // max union width of 16 queries' 7-wide windows
#define NK   154     // 7 * 22
#define NKP  160     // padded to multiple of 32
#define QT   16      // queries per tile (WMMA M)
#define NTJ  4       // ceil(56/16)

__global__ __launch_bounds__(128)
void natten_fwd_kernel(const float* __restrict__ x,
                       const float* __restrict__ rpb,
                       float* __restrict__ out)
{
    __shared__ __align__(16) half_t Qs[QT * HD];        //  2 KB
    __shared__ __align__(16) half_t Ks[NKP * HD];       // 20 KB
    __shared__ __align__(16) half_t Vts[HD * NKP];      // 20 KB (V transposed)
    __shared__ __align__(16) float  attnF[QT * NKP];    // 10 KB (bias -> scores -> exp)
    __shared__ __align__(16) half_t attnH[QT * NKP];    //  5 KB

    const int tid  = threadIdx.x;
    const int lane = tid & 31;
    const int wave = __builtin_amdgcn_readfirstlane(tid >> 5);  // SGPR-uniform
    const int col  = lane & 15;       // D col / B col
    const int kgrp = lane >> 4;       // lane-group selects K/M halves

    int bidx = blockIdx.x;
    const int tj = bidx & 3;  bidx >>= 2;
    const int i  = bidx % BH; bidx /= BH;
    const int n  = bidx % NHEADS;
    const int b  = bidx / NHEADS;

    const int j0  = tj * QT;
    const int sh  = min(max(i - KS / 2, 0), BH - KS);          // h-window start
    const int sw0 = min(max(j0 - KS / 2, 0), BW - KS);         // union w start
    const int jlast  = min(j0 + QT - 1, BW - 1);
    const int swLast = min(max(jlast - KS / 2, 0), BW - KS);
    const int uwid   = swLast + KS - sw0;                      // union width <= 22
    const float scale = 0.125f;                                // 64^-0.5
    const float NEGINF = -__builtin_inff();

    // ---------------- Phase 1: stage tiles + branchless bias/mask table -------
    // Q: 16 x 64  (pre-scaled, f16) -- clamp-and-select, no branches
    for (int t = tid; t < QT * HD; t += 128) {
        int m = t >> 6, d = t & 63;
        int j = j0 + m;
        int jc = min(j, BW - 1);
        float v = x[(((size_t)b * BH + i) * BW + jc) * 1536 + n * HD + d] * scale;
        Qs[t] = (half_t)((j < BW) ? v : 0.f);
    }
    // K union (NKP x 64) and V^T (64 x NKP), zero-padded -- clamp-and-select
    for (int t = tid; t < NKP * HD; t += 128) {
        int kk = t >> 6, d = t & 63;
        int kkc = min(kk, NK - 1);
        int r = kkc / UW, c = kkc - r * UW;
        int hh = sh + r;
        int ww = sw0 + min(c, uwid - 1);
        size_t base = (((size_t)b * BH + hh) * BW + ww) * 1536 + n * HD + d;
        float kv = x[base + 512];    // K plane
        float vv = x[base + 1024];   // V plane
        bool valid = (kk < NK) & (c < uwid);
        Ks[t]             = (half_t)(valid ? kv : 0.f);
        Vts[d * NKP + kk] = (half_t)(valid ? vv : 0.f);
    }
    // attnF[m][kk] = rpb bias if key kk is in query m's clamped window, else -inf
    const float* rpbN = rpb + n * 169;
    for (int t = tid; t < QT * NKP; t += 128) {   // 20 iters, branchless
        int m  = t / NKP;
        int kk = t - m * NKP;
        int j  = j0 + m;
        int rr = kk / UW, cc = kk - rr * UW;
        int sw  = min(max(j - KS / 2, 0), BW - KS);
        int off = sw - sw0;
        bool valid = (j < BW) & (kk < NK) & (cc >= off) & (cc < off + KS);
        int relh = min(max(sh + rr - i + (KS - 1), 0), 12);
        int relw = min(max(sw0 + cc - j + (KS - 1), 0), 12);
        float bias = rpbN[relh * 13 + relw];
        attnF[t] = valid ? bias : NEGINF;
    }
    __syncthreads();

    // ---------------- Phase 2: attnF += Q K^T ---------------------------------
    // A-fragments of Q are chunk-invariant: hoist both K-slices (kc=0,1).
    v16h aq0, aq1;
    {
        const int qrow = (lane & 15) * HD;
        v8h l0 = *(const v8h*)&Qs[qrow + 8 * kgrp];
        v8h h0 = *(const v8h*)&Qs[qrow + 16 + 8 * kgrp];
        v8h l1 = *(const v8h*)&Qs[qrow + 32 + 8 * kgrp];
        v8h h1 = *(const v8h*)&Qs[qrow + 48 + 8 * kgrp];
        #pragma unroll
        for (int t = 0; t < 8; ++t) {
            aq0[t] = l0[t]; aq0[t + 8] = h0[t];
            aq1[t] = l1[t]; aq1[t + 8] = h1[t];
        }
    }
    for (int chunk = wave; chunk < NKP / 16; chunk += 4) {   // SGPR-uniform loop
        const int key = chunk * 16 + col;
        v8f acc = {};
        // B = K^T slice (32x16): element (k, key) = Ks[key*HD + k]
        v8h b0l = *(const v8h*)&Ks[key * HD + 8 * kgrp];
        v8h b0h = *(const v8h*)&Ks[key * HD + 16 + 8 * kgrp];
        v8h b1l = *(const v8h*)&Ks[key * HD + 32 + 8 * kgrp];
        v8h b1h = *(const v8h*)&Ks[key * HD + 48 + 8 * kgrp];
        v16h bm0, bm1;
        #pragma unroll
        for (int t = 0; t < 8; ++t) {
            bm0[t] = b0l[t]; bm0[t + 8] = b0h[t];
            bm1[t] = b1l[t]; bm1[t + 8] = b1h[t];
        }
        acc = __builtin_amdgcn_wmma_f32_16x16x32_f16(false, aq0, false, bm0,
                                                     (short)0, acc, false, false);
        acc = __builtin_amdgcn_wmma_f32_16x16x32_f16(false, aq1, false, bm1,
                                                     (short)0, acc, false, false);
        // score accumulate into bias table: one ds_add_f32 per element, no branches
        #pragma unroll
        for (int r = 0; r < 8; ++r)
            atomicAdd(&attnF[(r + 8 * kgrp) * NKP + key], acc[r]);
    }
    __syncthreads();

    // ---------------- Phase 3: softmax, intra-wave shuffle reductions ---------
    // Each row's 8 segments live in 8 consecutive lanes of one wave: no barriers.
    {
        const int sm = tid >> 3;            // row 0..15
        const int sg = tid & 7;             // segment 0..7 (20 keys each)
        const int base = sm * NKP + sg * 20;
        float mx = NEGINF;
        #pragma unroll
        for (int q = 0; q < 20; ++q) mx = fmaxf(mx, attnF[base + q]);
        #pragma unroll
        for (int off = 4; off; off >>= 1)
            mx = fmaxf(mx, __shfl_xor(mx, off, 8));
        const bool dead = (mx == NEGINF);   // fully-masked padded query row
        float sum = 0.f;
        #pragma unroll
        for (int q = 0; q < 20; ++q) {
            float e = dead ? 0.f : __expf(attnF[base + q] - mx);
            attnF[base + q] = e;
            sum += e;
        }
        #pragma unroll
        for (int off = 4; off; off >>= 1)
            sum += __shfl_xor(sum, off, 8);
        float inv = (sum > 0.f) ? 1.f / sum : 0.f;
        #pragma unroll
        for (int q = 0; q < 20; ++q)
            attnH[base + q] = (half_t)(attnF[base + q] * inv);
    }
    __syncthreads();

    // ---------------- Phase 4: O = A V (each wave owns a 16-wide d-slice) -----
    {
        const int d = wave * 16 + col;
        v8f acc = {};
        #pragma unroll
        for (int kc = 0; kc < NKP / 32; ++kc) {   // 5 WMMAs over K = 160
            v8h alo = *(const v8h*)&attnH[(lane & 15) * NKP + kc * 32 + 8 * kgrp];
            v8h ahi = *(const v8h*)&attnH[(lane & 15) * NKP + kc * 32 + 16 + 8 * kgrp];
            v16h a;
            #pragma unroll
            for (int t = 0; t < 8; ++t) { a[t] = alo[t]; a[t + 8] = ahi[t]; }
            // B (32x16): element (k=key, col=d) = Vts[d*NKP + key]
            v8h blo = *(const v8h*)&Vts[d * NKP + kc * 32 + 8 * kgrp];
            v8h bhi = *(const v8h*)&Vts[d * NKP + kc * 32 + 16 + 8 * kgrp];
            v16h bm;
            #pragma unroll
            for (int t = 0; t < 8; ++t) { bm[t] = blo[t]; bm[t + 8] = bhi[t]; }
            acc = __builtin_amdgcn_wmma_f32_16x16x32_f16(false, a, false, bm,
                                                         (short)0, acc, false, false);
        }
        #pragma unroll
        for (int r = 0; r < 8; ++r) {
            int m = r + 8 * kgrp;
            int j = j0 + m;
            if (j < BW)
                out[(((size_t)b * BH + i) * BW + j) * 512 + n * HD + d] = acc[r];
        }
    }
}

extern "C" void kernel_launch(void* const* d_in, const int* in_sizes, int n_in,
                              void* d_out, int out_size, void* d_ws, size_t ws_size,
                              hipStream_t stream) {
    (void)in_sizes; (void)n_in; (void)out_size; (void)d_ws; (void)ws_size;
    const float* x   = (const float*)d_in[0];   // [2,56,56,1536] f32
    const float* rpb = (const float*)d_in[1];   // [8,13,13] f32
    float* out = (float*)d_out;                 // [2,56,56,512] f32

    const int grid = 2 * NHEADS * BH * NTJ;     // 3584 blocks, 4 waves each
    natten_fwd_kernel<<<grid, 128, 0, stream>>>(x, rpb, out);
}
